// KQV_28956669510232
// MI455X (gfx1250) — matrix operand order repaired and checked by reading, not compiled, not run on hardware
//
#include <hip/hip_runtime.h>
#include <hip/hip_bf16.h>

typedef _Float16 v16h __attribute__((ext_vector_type(16)));
typedef _Float16 v8h  __attribute__((ext_vector_type(8)));
typedef float    v8f  __attribute__((ext_vector_type(8)));
typedef float    v4f  __attribute__((ext_vector_type(4)));

#define BB 4
#define SS 2048
#define DD 1024

#define BM 128
#define BN 128
#define BK 32
#define KP 40   // padded LDS row stride (halves): 80B rows, 16B aligned, conflict-breaking

__device__ __forceinline__ v16h mk16(v8h lo, v8h hi) {
  return __builtin_shufflevector(lo, hi, 0,1,2,3,4,5,6,7,8,9,10,11,12,13,14,15);
}

// Generic tiled WMMA GEMM:  C[z] = A[z] (MxK) * B[z] + bias, optionally scaled.
//  TA/TB: float (converted to f16 on stage) or _Float16
//  BT   : true  -> B source is [K][N] row-major (transpose-staged into LDS [N][K])
//         false -> B source is [N][K] row-major (direct row copy; i.e. C = A * Bsrc^T)
//  TO   : _Float16 or float
// Software-pipelined: double-buffered LDS, global->reg prefetch overlaps WMMA,
// single barrier per K-step.
template <typename TA, typename TB, bool BT, typename TO, bool BIAS>
__global__ __launch_bounds__(256) void gemm_wmma(
    const TA* __restrict__ A, const TB* __restrict__ Bm,
    const float* __restrict__ bias, TO* __restrict__ C,
    int M, int N, int K, int lda, int ldb, int ldo,
    size_t bsA, size_t bsB, size_t bsC, float scale)
{
  __shared__ __align__(16) _Float16 As[2][BM][KP];
  __shared__ __align__(16) _Float16 Bs[2][BN][KP];

  A  += (size_t)blockIdx.z * bsA;
  Bm += (size_t)blockIdx.z * bsB;
  C  += (size_t)blockIdx.z * bsC;

  const int tid  = threadIdx.x;
  const int lane = tid & 31;
  const int wv   = tid >> 5;          // wave 0..7
  const int wr   = (wv & 3) * 32;     // wave M offset inside tile (4 waves down)
  const int wc   = (wv >> 2) * 64;    // wave N offset inside tile (2 waves across)
  const int lm   = lane & 15;
  const int lg   = lane >> 4;         // half-wave group

  const int tileN = blockIdx.x * BN;
  const int tileM = blockIdx.y * BM;

  v8f acc[2][4];
#pragma unroll
  for (int i = 0; i < 2; ++i)
#pragma unroll
    for (int j = 0; j < 4; ++j)
      acc[i][j] = (v8f)(0.0f);

  // ---------------- staging helpers (global->reg prefetch, reg->LDS store) --
  const int rA  = tid >> 1;           // A/[N][K]-B staging: 2 threads per row
  const int kA0 = (tid & 1) * 16;
  const int kB  = tid >> 3;           // [K][N]-B transpose staging
  const int nB0 = (tid & 7) * 16;

  v4f pa32[4]; v8h pa16[2];           // A prefetch regs (raw, convert at store)
  v4f pb32[4]; v8h pb16[2];           // B prefetch regs

  auto gload_A = [&](int kkx) {
    const TA* src = A + (size_t)(tileM + rA) * lda + (kkx + kA0);
    if constexpr (sizeof(TA) == 4) {
      pa32[0] = *(const v4f*)(src + 0);
      pa32[1] = *(const v4f*)(src + 4);
      pa32[2] = *(const v4f*)(src + 8);
      pa32[3] = *(const v4f*)(src + 12);
    } else {
      pa16[0] = *(const v8h*)(src);
      pa16[1] = *(const v8h*)(src + 8);
    }
  };
  auto sstore_A = [&](int buf) {
    v8h lo, hi;
    if constexpr (sizeof(TA) == 4) {
#pragma unroll
      for (int i = 0; i < 4; ++i) {
        lo[i]     = (_Float16)pa32[0][i];
        lo[4 + i] = (_Float16)pa32[1][i];
        hi[i]     = (_Float16)pa32[2][i];
        hi[4 + i] = (_Float16)pa32[3][i];
      }
    } else { lo = pa16[0]; hi = pa16[1]; }
    *(v8h*)&As[buf][rA][kA0]     = lo;
    *(v8h*)&As[buf][rA][kA0 + 8] = hi;
  };

  auto gload_B = [&](int kkx) {
    if constexpr (!BT) {
      const TB* src = Bm + (size_t)(tileN + rA) * ldb + (kkx + kA0);
      pb16[0] = *(const v8h*)(src);
      pb16[1] = *(const v8h*)(src + 8);
    } else {
      const TB* src = Bm + (size_t)(kkx + kB) * ldb + (tileN + nB0);
      if constexpr (sizeof(TB) == 4) {
        pb32[0] = *(const v4f*)(src + 0);
        pb32[1] = *(const v4f*)(src + 4);
        pb32[2] = *(const v4f*)(src + 8);
        pb32[3] = *(const v4f*)(src + 12);
      } else {
        pb16[0] = *(const v8h*)(src);
        pb16[1] = *(const v8h*)(src + 8);
      }
    }
  };
  auto sstore_B = [&](int buf) {
    if constexpr (!BT) {
      *(v8h*)&Bs[buf][rA][kA0]     = pb16[0];
      *(v8h*)&Bs[buf][rA][kA0 + 8] = pb16[1];
    } else {
      _Float16 h[16];
      if constexpr (sizeof(TB) == 4) {
#pragma unroll
        for (int i = 0; i < 4; ++i) {
          h[i]      = (_Float16)pb32[0][i];
          h[4 + i]  = (_Float16)pb32[1][i];
          h[8 + i]  = (_Float16)pb32[2][i];
          h[12 + i] = (_Float16)pb32[3][i];
        }
      } else {
#pragma unroll
        for (int i = 0; i < 8; ++i) { h[i] = pb16[0][i]; h[8 + i] = pb16[1][i]; }
      }
#pragma unroll
      for (int i = 0; i < 16; ++i) Bs[buf][nB0 + i][kB] = h[i];
    }
  };

  // ---------------- fragment loads + 8 WMMAs on one LDS buffer -------------
  auto compute = [&](int buf) {
    // B fragment (32x16, contiguous k = 16*lg + s): lane lm = N, lg selects K 0-15 / 16-31,
    // contiguous in [N][K] staging -> two ds_load_b128 per fragment.
    v16h bfr[4];
#pragma unroll
    for (int j = 0; j < 4; ++j) {
      const int bn = wc + j * 16 + lm;
      bfr[j] = mk16(*(const v8h*)&Bs[buf][bn][lg * 16],
                    *(const v8h*)&Bs[buf][bn][lg * 16 + 8]);
    }
#pragma unroll
    for (int i = 0; i < 2; ++i) {
      // A fragment (16x32): lane lm = M row; lanes 0-15 hold K 0-15,
      // lanes 16-31 hold K 16-31 -> two ds_load_b128 per fragment.
      const int ar = wr + i * 16 + lm;
      v16h a = mk16(*(const v8h*)&As[buf][ar][lg * 16],
                    *(const v8h*)&As[buf][ar][lg * 16 + 8]);
#pragma unroll
      for (int j = 0; j < 4; ++j) {
        acc[i][j] = __builtin_amdgcn_wmma_f32_16x16x32_f16(
            false, a, false, bfr[j], (short)0, acc[i][j], false, false);
      }
    }
  };

  // ---------------- pipelined main loop ------------------------------------
  gload_A(0);
  gload_B(0);
  sstore_A(0);
  sstore_B(0);
  __syncthreads();

  const int nsteps = K / BK;
  for (int s = 0; s < nsteps; ++s) {
    const int  cur  = s & 1;
    const bool more = (s + 1) < nsteps;
    if (more) {                 // issue next tile's global loads first ...
      gload_A((s + 1) * BK);
      gload_B((s + 1) * BK);
    }
    compute(cur);               // ... 8 WMMAs overlap with loads in flight
    if (more) {                 // then drain loads into the other buffer
      sstore_A(cur ^ 1);
      sstore_B(cur ^ 1);
    }
    __syncthreads();            // single barrier per K-step (buffer flip)
  }

  // ---- writeout (C layout: VGPR r -> M = 8*lg + r, N = lm) ----
#pragma unroll
  for (int i = 0; i < 2; ++i) {
#pragma unroll
    for (int j = 0; j < 4; ++j) {
      const int n = tileN + wc + j * 16 + lm;
      float badd = 0.0f;
      if constexpr (BIAS) badd = bias[n];
      const int mbase = tileM + wr + i * 16 + lg * 8;
#pragma unroll
      for (int r = 0; r < 8; ++r) {
        float v = acc[i][j][r] * scale + badd;
        C[(size_t)(mbase + r) * ldo + n] = (TO)v;
      }
    }
  }
}

// In-place row softmax over S=2048 f16 scores; one 256-thread block per row.
__global__ __launch_bounds__(256) void softmax_inplace(_Float16* __restrict__ sc)
{
  __shared__ float red[256];
  const int tid = threadIdx.x;
  _Float16* row = sc + (size_t)blockIdx.x * SS;

  v8h x = *(const v8h*)(row + tid * 8);
  float v[8];
  float m = -3.0e38f;
#pragma unroll
  for (int i = 0; i < 8; ++i) { v[i] = (float)x[i]; m = fmaxf(m, v[i]); }

  red[tid] = m; __syncthreads();
  for (int s = 128; s > 0; s >>= 1) {
    if (tid < s) red[tid] = fmaxf(red[tid], red[tid + s]);
    __syncthreads();
  }
  m = red[0]; __syncthreads();

  float sum = 0.0f;
#pragma unroll
  for (int i = 0; i < 8; ++i) { v[i] = __expf(v[i] - m); sum += v[i]; }

  red[tid] = sum; __syncthreads();
  for (int s = 128; s > 0; s >>= 1) {
    if (tid < s) red[tid] += red[tid + s];
    __syncthreads();
  }
  const float inv = 1.0f / red[0];

  v8h y;
#pragma unroll
  for (int i = 0; i < 8; ++i) y[i] = (_Float16)(v[i] * inv);
  *(v8h*)(row + tid * 8) = y;
}

extern "C" void kernel_launch(void* const* d_in, const int* in_sizes, int n_in,
                              void* d_out, int out_size, void* d_ws, size_t ws_size,
                              hipStream_t stream) {
  (void)in_sizes; (void)n_in; (void)out_size; (void)ws_size;

  const float* x  = (const float*)d_in[0];
  const float* Wk = (const float*)d_in[1];
  const float* bk = (const float*)d_in[2];
  const float* Wq = (const float*)d_in[3];
  const float* bq = (const float*)d_in[4];
  const float* Wv = (const float*)d_in[5];
  const float* bv = (const float*)d_in[6];
  float* out = (float*)d_out;

  const size_t BSD = (size_t)BB * SS * DD;       // 8,388,608 halves each
  _Float16* Qh = (_Float16*)d_ws;
  _Float16* Kh = Qh + BSD;
  _Float16* Vh = Kh + BSD;
  _Float16* P  = Vh + BSD;                       // B*S*S halves (scores -> probs)

  dim3 blk(256);

  // 1) Projections: [B*S, D] = [B*S, D] x [D, D] + bias, f32 in -> f16 out
  dim3 gp(DD / BN, (BB * SS) / BM, 1);
  gemm_wmma<float, float, true, _Float16, true><<<gp, blk, 0, stream>>>(
      x, Wq, bq, Qh, BB * SS, DD, DD, DD, DD, DD, 0, 0, 0, 1.0f);
  gemm_wmma<float, float, true, _Float16, true><<<gp, blk, 0, stream>>>(
      x, Wk, bk, Kh, BB * SS, DD, DD, DD, DD, DD, 0, 0, 0, 1.0f);
  gemm_wmma<float, float, true, _Float16, true><<<gp, blk, 0, stream>>>(
      x, Wv, bv, Vh, BB * SS, DD, DD, DD, DD, DD, 0, 0, 0, 1.0f);

  // 2) Scores: per batch  S[b] = Q[b] x K[b]^T * (1/sqrt(D));  B source is [N][K]
  dim3 gs(SS / BN, SS / BM, BB);
  gemm_wmma<_Float16, _Float16, false, _Float16, false><<<gs, blk, 0, stream>>>(
      Qh, Kh, nullptr, P, SS, SS, DD, DD, DD, SS,
      (size_t)SS * DD, (size_t)SS * DD, (size_t)SS * SS, 0.03125f /* 1/32 */);

  // 3) Softmax over keys, in place on P
  softmax_inplace<<<dim3(BB * SS), blk, 0, stream>>>(P);

  // 4) Output: per batch  O[b] = P[b] x V[b];  B source is [K][N], f32 out
  dim3 go(DD / BN, SS / BM, BB);
  gemm_wmma<_Float16, _Float16, true, float, false><<<go, blk, 0, stream>>>(
      P, Vh, nullptr, out, SS, DD, SS, SS, DD, DD,
      (size_t)SS * SS, (size_t)SS * DD, (size_t)SS * DD, 1.0f);
}
